// HybridLSTMODE_33260226740404
// MI455X (gfx1250) — matrix operand order repaired and checked by reading, not compiled, not run on hardware
//
#include <hip/hip_runtime.h>
#include <hip/hip_bf16.h>
#include <math.h>

#define LATENT 64
#define HIDDEN 128
#define N_IN   21
#define N_OUT  20
#define BATCH  512
#define SEQ    1024

typedef __attribute__((ext_vector_type(16))) _Float16 v16h;
typedef __attribute__((ext_vector_type(8)))  float    v8f;

// ---------------------------------------------------------------------------
// WMMA helper: D = A(16x32 f16) x B(32x16 f16) + C(16x16 f32)
// ---------------------------------------------------------------------------
static __device__ __forceinline__ v8f wmma_f16(v16h a, v16h b, v8f c) {
  return __builtin_amdgcn_wmma_f32_16x16x32_f16(false, a, false, b, (short)0, c,
                                                false, false);
}

// ---------------------------------------------------------------------------
// Branch-free activations: v_tanh_f32 (confirmed on gfx1250) and
// v_exp_f32 + v_rcp_f32 sigmoid.
// ---------------------------------------------------------------------------
#define LOG2E 1.4426950408889634f

static __device__ __forceinline__ float fast_sigmoid(float v) {
  return __builtin_amdgcn_rcpf(1.0f + __builtin_amdgcn_exp2f(v * -LOG2E));
}

static __device__ __forceinline__ float fast_tanh(float v) {
#if __has_builtin(__builtin_amdgcn_tanhf)
  return __builtin_amdgcn_tanhf(v);
#else
  return 1.0f - 2.0f * __builtin_amdgcn_rcpf(
                           1.0f + __builtin_amdgcn_exp2f(v * (2.0f * LOG2E)));
#endif
}

// ---------------------------------------------------------------------------
// B-fragment loader (global f32 weights W[N][ldk], B[k][n] = W[n][k]).
// wave32 B layout for 16x16x32 f16: column n=n0+(lane&15); elems 0-7 =
// K=k0+kb+0..7, elems 8-15 = K=k0+kb+16..23, kb = (lane>=16 ? 8 : 0).
// n/k guards allow zero-padded tiles (K=21->32, N=20->32).
// ---------------------------------------------------------------------------
static __device__ __forceinline__ v16h load_bfrag(const float* __restrict__ W,
                                                  int ldk, int n0, int k0,
                                                  int nvalid, int kvalid,
                                                  int lane) {
  int n  = n0 + (lane & 15);
  int kk = k0 + ((lane >> 4) << 3);
  const float* p = W + (size_t)n * ldk + kk;
  v16h r;
#pragma unroll
  for (int e = 0; e < 16; ++e) {
    int o = e + ((e >> 3) << 3);
    float v = (n < nvalid && (kk + o) < kvalid) ? p[o] : 0.0f;
    r[e] = (_Float16)v;
  }
  return r;
}

// ---------------------------------------------------------------------------
// A-fragment loader from LDS f16 row-major [16][ldk].
// wave32 A layout: lane holds row m=lane&15; elems 0-7 = K=k0+kb+0..7,
// elems 8-15 = K=k0+kb+16..23, kb = (lane>=16 ? 8 : 0).
// ---------------------------------------------------------------------------
static __device__ __forceinline__ v16h afrag_lds(const _Float16* A, int ldk,
                                                 int k0, int lane) {
  int m  = lane & 15;
  int kb = k0 + ((lane >> 4) << 3);
  const _Float16* p = A + m * ldk + kb;
  v16h r;
#pragma unroll
  for (int e = 0; e < 8; ++e) r[e] = p[e];
#pragma unroll
  for (int e = 0; e < 8; ++e) r[8 + e] = p[16 + e];
  return r;
}

// A-fragment from global f32 rows (decoder reads zs), same striping + cvt.
static __device__ __forceinline__ v16h afrag_g32(const float* __restrict__ A,
                                                 size_t row0, int ldk, int k0,
                                                 int lane) {
  int m  = lane & 15;
  int kb = k0 + ((lane >> 4) << 3);
  const float* p = A + (row0 + m) * ldk + kb;
  v16h r;
#pragma unroll
  for (int e = 0; e < 8; ++e) r[e] = (_Float16)p[e];
#pragma unroll
  for (int e = 0; e < 8; ++e) r[8 + e] = (_Float16)p[16 + e];
  return r;
}

// A-fragment for x[b][t][0:21] (K padded 21->32 with zeros).
static __device__ __forceinline__ v16h afrag_x(const float* __restrict__ x,
                                               int b0, int t, int lane) {
  int m  = lane & 15;
  int kb = (lane >> 4) << 3;
  const float* p = x + ((size_t)(b0 + m) * SEQ + t) * N_IN;
  v16h r;
#pragma unroll
  for (int e = 0; e < 8; ++e) {
    int k = kb + e;
    r[e] = (k < N_IN) ? (_Float16)p[k] : (_Float16)0.0f;
  }
#pragma unroll
  for (int e = 0; e < 8; ++e) {
    int k = kb + 16 + e;
    r[8 + e] = (k < N_IN) ? (_Float16)p[k] : (_Float16)0.0f;
  }
  return r;
}

// C-tile (v8f) element mapping: elem r -> row m = r + 8*(lane>=16), col lane&15.
// Activation selected at compile time; callers branch ONCE per wave (uniform)
// instead of a per-element exec-mask diamond.
template <bool TANH>
static __device__ __forceinline__ void act_store_f32(float* dst, int ldn,
                                                     const float* bias, v8f acc,
                                                     int n0, int lane) {
  int n  = n0 + (lane & 15);
  int mb = (lane >> 4) << 3;
  float b = bias[n];
#pragma unroll
  for (int r = 0; r < 8; ++r) {
    float v = acc[r] + b;
    v = TANH ? fast_tanh(v) : fast_sigmoid(v);
    dst[(mb + r) * ldn + n] = v;
  }
}

static __device__ __forceinline__ void tanh_store_f16(_Float16* dst, int ldn,
                                                      const float* bias, v8f acc,
                                                      int n0, int lane) {
  int n  = n0 + (lane & 15);
  int mb = (lane >> 4) << 3;
  float b = bias[n];
#pragma unroll
  for (int r = 0; r < 8; ++r)
    dst[(mb + r) * ldn + n] = (_Float16)fast_tanh(acc[r] + b);
}

// ===========================================================================
// Fused sequential kernel: LSTM0 -> LSTM1 -> gated latent-ODE midpoint scan.
// One workgroup per 16-row batch tile; 8 waves split N-tiles; all weight
// B-fragments live in VGPRs for the whole sequence; state lives in LDS.
// ===========================================================================
__global__ __launch_bounds__(256, 1)
void lstm_ode_scan_kernel(const float* __restrict__ x,
                          const float* __restrict__ Wih0,
                          const float* __restrict__ Whh0,
                          const float* __restrict__ bih0,
                          const float* __restrict__ bhh0,
                          const float* __restrict__ Wih1,
                          const float* __restrict__ Whh1,
                          const float* __restrict__ bih1,
                          const float* __restrict__ bhh1,
                          const float* __restrict__ Wg,
                          const float* __restrict__ bg,
                          const float* __restrict__ Wo1,
                          const float* __restrict__ bo1,
                          const float* __restrict__ Wo2,
                          const float* __restrict__ bo2,
                          float* __restrict__ lat_out) {
  __shared__ _Float16 sH0[16 * 64];    // layer0 hidden (A operand, f16)
  __shared__ _Float16 sHZ[16 * 128];   // [h1 | z_ode] concat (A for gate)
  __shared__ _Float16 sZg[16 * 64];    // gated z (A for Wo1 pass 1)
  __shared__ _Float16 sZ2[16 * 64];    // z + 0.5*k1 (A for Wo1 pass 2)
  __shared__ _Float16 sA[16 * 128];    // tanh hidden of ODE net (A for Wo2)
  __shared__ float sG[16 * 256];       // activated LSTM gates scratch
  __shared__ float sC0[16 * 64], sC1[16 * 64];
  __shared__ float sZo[16 * 64];       // z_ode carry (f32)
  __shared__ float sZf[16 * 64];       // gated z (f32)
  __shared__ float sB0[256], sB1[256], sBg[64], sBo1[128], sBo2[64];

  const int tid  = threadIdx.x;
  const int lane = tid & 31;
  const int wv   = tid >> 5;
  const int b0   = blockIdx.x * 16;

  // ---- init biases and zero state ----
  if (tid < 256) { sB0[tid] = bih0[tid] + bhh0[tid]; sB1[tid] = bih1[tid] + bhh1[tid]; }
  if (tid < 64)  sBg[tid]  = bg[tid];
  if (tid < 128) sBo1[tid] = bo1[tid];
  if (tid < 64)  sBo2[tid] = bo2[tid];
  for (int i = tid; i < 16 * 64; i += 256) {
    sC0[i] = 0.0f; sC1[i] = 0.0f; sZo[i] = 0.0f;
    sH0[i] = (_Float16)0.0f;
  }
  for (int i = tid; i < 16 * 128; i += 256) sHZ[i] = (_Float16)0.0f;

  // ---- preload all weight B-fragments into VGPRs (reused SEQ times) ----
  const int t0 = wv * 2, t1 = wv * 2 + 1;  // this wave's N-tiles of the 256-wide gates
  // Both tiles of a wave belong to the same LSTM gate: t0>>2 == t1>>2.
  const bool gate_is_tanh = ((t0 >> 2) == 2);

  v16h fX0 = load_bfrag(Wih0, N_IN, t0 * 16, 0, 256, N_IN, lane);
  v16h fX1 = load_bfrag(Wih0, N_IN, t1 * 16, 0, 256, N_IN, lane);
  v16h fH0[2][2], fI1[2][2], fH1[2][2];
#pragma unroll
  for (int c = 0; c < 2; ++c) {
    fH0[0][c] = load_bfrag(Whh0, 64, t0 * 16, c * 32, 256, 64, lane);
    fH0[1][c] = load_bfrag(Whh0, 64, t1 * 16, c * 32, 256, 64, lane);
    fI1[0][c] = load_bfrag(Wih1, 64, t0 * 16, c * 32, 256, 64, lane);
    fI1[1][c] = load_bfrag(Wih1, 64, t1 * 16, c * 32, 256, 64, lane);
    fH1[0][c] = load_bfrag(Whh1, 64, t0 * 16, c * 32, 256, 64, lane);
    fH1[1][c] = load_bfrag(Whh1, 64, t1 * 16, c * 32, 256, 64, lane);
  }
  v16h fG[4] = {}, fO2[4] = {};
  if (wv < 4) {
#pragma unroll
    for (int c = 0; c < 4; ++c)
      fG[c] = load_bfrag(Wg, 128, wv * 16, c * 32, 64, 128, lane);
  } else {
#pragma unroll
    for (int c = 0; c < 4; ++c)
      fO2[c] = load_bfrag(Wo2, 128, (wv - 4) * 16, c * 32, 64, 128, lane);
  }
  v16h fO1[2];
#pragma unroll
  for (int c = 0; c < 2; ++c)
    fO1[c] = load_bfrag(Wo1, 64, wv * 16, c * 32, 128, 64, lane);

  __syncthreads();

  for (int t = 0; t < SEQ; ++t) {
    // ============ Phase A: LSTM layer 0 gates ============
    {
      v16h ax = afrag_x(x, b0, t, lane);
      if (wv == 0 && t + 1 < SEQ) {
        const float* pn = x + ((size_t)(b0 + (lane & 15)) * SEQ + (t + 1)) * N_IN;
        __builtin_prefetch(pn, 0, 1);  // global_prefetch_b8 for next timestep
      }
      v16h ah0 = afrag_lds(sH0, 64, 0, lane);
      v16h ah1 = afrag_lds(sH0, 64, 32, lane);
      v8f a0 = {}, a1 = {};
      a0 = wmma_f16(ax, fX0, a0);
      a0 = wmma_f16(ah0, fH0[0][0], a0);
      a0 = wmma_f16(ah1, fH0[0][1], a0);
      a1 = wmma_f16(ax, fX1, a1);
      a1 = wmma_f16(ah0, fH0[1][0], a1);
      a1 = wmma_f16(ah1, fH0[1][1], a1);
      if (gate_is_tanh) {  // one uniform branch per wave, not per element
        act_store_f32<true>(sG, 256, sB0, a0, t0 * 16, lane);
        act_store_f32<true>(sG, 256, sB0, a1, t1 * 16, lane);
      } else {
        act_store_f32<false>(sG, 256, sB0, a0, t0 * 16, lane);
        act_store_f32<false>(sG, 256, sB0, a1, t1 * 16, lane);
      }
    }
    __syncthreads();
    // elementwise c/h update, layer 0
    for (int i = tid; i < 16 * 64; i += 256) {
      int m = i >> 6, j = i & 63;
      float iv = sG[m * 256 + j];
      float fv = sG[m * 256 + 64 + j];
      float gv = sG[m * 256 + 128 + j];
      float ov = sG[m * 256 + 192 + j];
      float c = fv * sC0[i] + iv * gv;
      sC0[i] = c;
      sH0[i] = (_Float16)(ov * fast_tanh(c));
    }
    __syncthreads();

    // ============ Phase B: LSTM layer 1 gates ============
    {
      v16h ai0 = afrag_lds(sH0, 64, 0, lane);
      v16h ai1 = afrag_lds(sH0, 64, 32, lane);
      v16h ar0 = afrag_lds(sHZ, 128, 0, lane);   // h1 cols 0-63
      v16h ar1 = afrag_lds(sHZ, 128, 32, lane);
      v8f a0 = {}, a1 = {};
      a0 = wmma_f16(ai0, fI1[0][0], a0);
      a0 = wmma_f16(ai1, fI1[0][1], a0);
      a0 = wmma_f16(ar0, fH1[0][0], a0);
      a0 = wmma_f16(ar1, fH1[0][1], a0);
      a1 = wmma_f16(ai0, fI1[1][0], a1);
      a1 = wmma_f16(ai1, fI1[1][1], a1);
      a1 = wmma_f16(ar0, fH1[1][0], a1);
      a1 = wmma_f16(ar1, fH1[1][1], a1);
      if (gate_is_tanh) {
        act_store_f32<true>(sG, 256, sB1, a0, t0 * 16, lane);
        act_store_f32<true>(sG, 256, sB1, a1, t1 * 16, lane);
      } else {
        act_store_f32<false>(sG, 256, sB1, a0, t0 * 16, lane);
        act_store_f32<false>(sG, 256, sB1, a1, t1 * 16, lane);
      }
    }
    __syncthreads();
    // elementwise c/h update, layer 1 (h1 = lstm_out[t]); seed z_ode at t==0
    for (int i = tid; i < 16 * 64; i += 256) {
      int m = i >> 6, j = i & 63;
      float iv = sG[m * 256 + j];
      float fv = sG[m * 256 + 64 + j];
      float gv = sG[m * 256 + 128 + j];
      float ov = sG[m * 256 + 192 + j];
      float c = fv * sC1[i] + iv * gv;
      sC1[i] = c;
      float h = ov * fast_tanh(c);
      sHZ[m * 128 + j] = (_Float16)h;
      if (t == 0) {
        sZo[i] = h;
        sHZ[m * 128 + 64 + j] = (_Float16)h;
      }
    }
    __syncthreads();

    // ============ Phase C: gate g = sigmoid([h1|z_ode] Wg^T + bg); blend ====
    if (wv < 4) {
      v8f acc = {};
#pragma unroll
      for (int c = 0; c < 4; ++c)
        acc = wmma_f16(afrag_lds(sHZ, 128, c * 32, lane), fG[c], acc);
      int n  = wv * 16 + (lane & 15);
      int mb = (lane >> 4) << 3;
#pragma unroll
      for (int r = 0; r < 8; ++r) {
        int m = mb + r;
        float g  = fast_sigmoid(acc[r] + sBg[n]);
        float hv = (float)sHZ[m * 128 + n];
        float z  = g * hv + (1.0f - g) * sZo[m * 64 + n];
        sZf[m * 64 + n] = z;
        sZg[m * 64 + n] = (_Float16)z;
        lat_out[((size_t)(b0 + m) * SEQ + t) * LATENT + n] = z;  // latent_traj
      }
    }
    __syncthreads();

    // ============ Phase D: midpoint ODE step ============
    // D1: a1 = tanh(z Wo1^T + bo1)
    {
      v8f acc = {};
      acc = wmma_f16(afrag_lds(sZg, 64, 0, lane), fO1[0], acc);
      acc = wmma_f16(afrag_lds(sZg, 64, 32, lane), fO1[1], acc);
      tanh_store_f16(sA, 128, sBo1, acc, wv * 16, lane);
    }
    __syncthreads();
    // D2: k1 = a1 Wo2^T + bo2 ; z2 = z + 0.5*k1
    if (wv >= 4) {
      v8f acc = {};
#pragma unroll
      for (int c = 0; c < 4; ++c)
        acc = wmma_f16(afrag_lds(sA, 128, c * 32, lane), fO2[c], acc);
      int n  = (wv - 4) * 16 + (lane & 15);
      int mb = (lane >> 4) << 3;
#pragma unroll
      for (int r = 0; r < 8; ++r) {
        int m = mb + r;
        float k1 = acc[r] + sBo2[n];
        sZ2[m * 64 + n] = (_Float16)(sZf[m * 64 + n] + 0.5f * k1);
      }
    }
    __syncthreads();
    // D3: a2 = tanh(z2 Wo1^T + bo1)
    {
      v8f acc = {};
      acc = wmma_f16(afrag_lds(sZ2, 64, 0, lane), fO1[0], acc);
      acc = wmma_f16(afrag_lds(sZ2, 64, 32, lane), fO1[1], acc);
      tanh_store_f16(sA, 128, sBo1, acc, wv * 16, lane);
    }
    __syncthreads();
    // D4: z_next = z + a2 Wo2^T + bo2 (new z_ode carry)
    if (wv >= 4) {
      v8f acc = {};
#pragma unroll
      for (int c = 0; c < 4; ++c)
        acc = wmma_f16(afrag_lds(sA, 128, c * 32, lane), fO2[c], acc);
      int n  = (wv - 4) * 16 + (lane & 15);
      int mb = (lane >> 4) << 3;
#pragma unroll
      for (int r = 0; r < 8; ++r) {
        int m = mb + r;
        float zn = sZf[m * 64 + n] + acc[r] + sBo2[n];
        sZo[m * 64 + n] = zn;
        sHZ[m * 128 + 64 + n] = (_Float16)zn;
      }
    }
    __syncthreads();
  }
}

// ===========================================================================
// Decoder: y = sigmoid(tanh(zs Wd1^T + bd1) Wd2^T + bd2), fully parallel.
// 1 row-tile (16 rows of flat [B*S]) per wave; Wd fragments in VGPRs.
// ===========================================================================
__global__ __launch_bounds__(256, 1)
void decoder_kernel(const float* __restrict__ zs,
                    const float* __restrict__ Wd1,
                    const float* __restrict__ bd1,
                    const float* __restrict__ Wd2,
                    const float* __restrict__ bd2,
                    float* __restrict__ y) {
  __shared__ _Float16 sHd[8][16 * 128];
  __shared__ float sBd1[128], sBd2[32];

  const int tid  = threadIdx.x;
  const int lane = tid & 31;
  const int wv   = tid >> 5;
  if (tid < 128) sBd1[tid] = bd1[tid];
  if (tid < 32)  sBd2[tid] = (tid < N_OUT) ? bd2[tid] : 0.0f;

  v16h fD1[8][2], fD2[2][4];
#pragma unroll
  for (int nt = 0; nt < 8; ++nt)
#pragma unroll
    for (int c = 0; c < 2; ++c)
      fD1[nt][c] = load_bfrag(Wd1, 64, nt * 16, c * 32, 128, 64, lane);
#pragma unroll
  for (int nt = 0; nt < 2; ++nt)
#pragma unroll
    for (int c = 0; c < 4; ++c)
      fD2[nt][c] = load_bfrag(Wd2, 128, nt * 16, c * 32, N_OUT, 128, lane);
  __syncthreads();

  const size_t tile = (size_t)blockIdx.x * 8 + wv;  // BATCH*SEQ/16 tiles total
  const size_t r0   = tile * 16;
  _Float16* hd = sHd[wv];

  v16h az0 = afrag_g32(zs, r0, LATENT, 0, lane);
  v16h az1 = afrag_g32(zs, r0, LATENT, 32, lane);

#pragma unroll
  for (int nt = 0; nt < 8; ++nt) {
    v8f acc = {};
    acc = wmma_f16(az0, fD1[nt][0], acc);
    acc = wmma_f16(az1, fD1[nt][1], acc);
    tanh_store_f16(hd, 128, sBd1, acc, nt * 16, lane);
  }
  // per-wave LDS RAW: compiler inserts the DScnt wait before the reads below
#pragma unroll
  for (int nt = 0; nt < 2; ++nt) {
    v8f acc = {};
#pragma unroll
    for (int c = 0; c < 4; ++c)
      acc = wmma_f16(afrag_lds(hd, 128, c * 32, lane), fD2[nt][c], acc);
    int n  = nt * 16 + (lane & 15);
    int mb = (lane >> 4) << 3;
#pragma unroll
    for (int r = 0; r < 8; ++r) {
      if (n < N_OUT)
        y[(r0 + mb + r) * N_OUT + n] = fast_sigmoid(acc[r] + sBd2[n]);
    }
  }
}

// ===========================================================================
extern "C" void kernel_launch(void* const* d_in, const int* in_sizes, int n_in,
                              void* d_out, int out_size, void* d_ws, size_t ws_size,
                              hipStream_t stream) {
  const float* x    = (const float*)d_in[0];
  const float* Wih0 = (const float*)d_in[1];
  const float* Whh0 = (const float*)d_in[2];
  const float* bih0 = (const float*)d_in[3];
  const float* bhh0 = (const float*)d_in[4];
  const float* Wih1 = (const float*)d_in[5];
  const float* Whh1 = (const float*)d_in[6];
  const float* bih1 = (const float*)d_in[7];
  const float* bhh1 = (const float*)d_in[8];
  const float* Wg   = (const float*)d_in[9];
  const float* bg   = (const float*)d_in[10];
  const float* Wo1  = (const float*)d_in[11];
  const float* bo1  = (const float*)d_in[12];
  const float* Wo2  = (const float*)d_in[13];
  const float* bo2  = (const float*)d_in[14];
  const float* Wd1  = (const float*)d_in[15];
  const float* bd1  = (const float*)d_in[16];
  const float* Wd2  = (const float*)d_in[17];
  const float* bd2  = (const float*)d_in[18];

  float* y_out   = (float*)d_out;                                  // [B,S,20]
  float* lat_out = (float*)d_out + (size_t)BATCH * SEQ * N_OUT;    // [B,S,64]

  // Fused sequential scan: one 16-row batch tile per workgroup.
  lstm_ode_scan_kernel<<<BATCH / 16, 256, 0, stream>>>(
      x, Wih0, Whh0, bih0, bhh0, Wih1, Whh1, bih1, bhh1,
      Wg, bg, Wo1, bo1, Wo2, bo2, lat_out);

  // Fully parallel decoder over all B*S latents (reads lat_out in d_out).
  decoder_kernel<<<(BATCH * SEQ / 16) / 8, 256, 0, stream>>>(
      lat_out, Wd1, bd1, Wd2, bd2, y_out);
}
